// Model_51685636440502
// MI455X (gfx1250) — matrix operand order repaired and checked by reading, not compiled, not run on hardware
//
#include <hip/hip_runtime.h>
#include <math.h>
#include <stdint.h>

typedef _Float16 half16 __attribute__((ext_vector_type(16)));
typedef float    float8 __attribute__((ext_vector_type(8)));

#define BB   128
#define LL   720
#define LP   736      // 720 padded to multiple of 32
#define CC   321
#define PP   336
#define MM   512
#define HID_ 128
#define NN   (BB * CC)        // 41088 = 1284 * 32
#define NT16 (NN / 16)        // 2568
#define NT32 (NN / 32)        // 1284
#define CGRP 21               // ceil(321/16)

// ---------------- async global -> LDS staging (CDNA5 path) -----------------
// Inline asm bypasses the builtin's CUDA-address-space parameter types.
// LDS operand: generic LDS pointer's low 32 bits are the wave-relative LDS
// byte address (flat aperture rule), which is what VDST expects.

__device__ __forceinline__ void copy16_g2l(_Float16* lds_dst, const _Float16* gsrc) {
    uint32_t lds_off = (uint32_t)(uintptr_t)lds_dst;
    asm volatile("global_load_async_to_lds_b128 %0, %1, off"
                 :: "v"(lds_off), "v"(gsrc)
                 : "memory");
}

__device__ __forceinline__ void staging_fence() {
    asm volatile("s_wait_asynccnt 0" ::: "memory");
    __syncthreads();
}

// ---------------- WMMA fragment loaders (CDNA5 16x16x32 f16 layouts) -------

// A: 16x32 f16, row-major source with leading dim `ld` (in halves).
// half e of lane: K = k0 + 16*(lane>=16) + e; M = row0 + lane&15
__device__ __forceinline__ half16 load_a_frag(const _Float16* A, int ld,
                                              int row0, int k0, int lane) {
    const _Float16* p = A + (size_t)(row0 + (lane & 15)) * ld
                          + k0 + ((lane >> 4) << 4);
    half16 a;
#pragma unroll
    for (int e = 0; e < 8; ++e) a[e] = p[e];
#pragma unroll
    for (int e = 0; e < 8; ++e) a[e + 8] = p[e + 8];
    return a;
}

// B: 32x16 f16 where B(k,n) = W[n*ld + k]  (row-major [Ncols, K] weight, transposed use)
__device__ __forceinline__ half16 load_b_frag_T(const _Float16* W, int ld,
                                                int n0, int k0, int lane) {
    const _Float16* p = W + (size_t)(n0 + (lane & 15)) * ld
                          + k0 + ((lane >> 4) << 4);
    half16 b;
#pragma unroll
    for (int e = 0; e < 16; ++e) b[e] = p[e];
    return b;
}

// B: 32x16 f16 where B(k,n) = M[k*ld + n]  (row-major [K, Ncols] matrix)
__device__ __forceinline__ half16 load_b_frag_N(const _Float16* Mm, int ld,
                                                int k0, int n0, int lane) {
    const _Float16* p = Mm + (size_t)(k0 + ((lane >> 4) << 4)) * ld
                           + n0 + (lane & 15);
    half16 b;
#pragma unroll
    for (int e = 0; e < 16; ++e) b[e] = p[(size_t)e * ld];
    return b;
}

__device__ __forceinline__ float fast_sigmoid(float x) {
    return __builtin_amdgcn_rcpf(1.f + __expf(-x));
}

// ---------------- weight conversion (f32 -> f16, optional K padding) -------

__global__ void pad_convert_f16_kernel(const float* __restrict__ src,
                                       _Float16* __restrict__ dst,
                                       int rows, int K, int Kp) {
    int idx = blockIdx.x * blockDim.x + threadIdx.x;
    if (idx >= rows * Kp) return;
    int r = idx / Kp, k = idx - r * Kp;
    dst[idx] = (k < K) ? (_Float16)src[(size_t)r * K + k] : (_Float16)0.f;
}

// ---------------- series decomposition (25-tap edge-clamped moving avg) ----

__global__ __launch_bounds__(256) void decompose_kernel(
    const float* __restrict__ x,       // [BB, LL, CC]
    _Float16* __restrict__ seas,       // [NN, LP]
    _Float16* __restrict__ trend) {    // [NN, LP]
    __shared__ float xs[16 * 725];     // odd stride -> conflict-free
    const int tid = threadIdx.x;
    const int b  = blockIdx.x / CGRP;
    const int c0 = (blockIdx.x % CGRP) * 16;

    for (int idx = tid; idx < 16 * LL; idx += 256) {
        int c_off = idx & 15, l = idx >> 4;
        int c = c0 + c_off;
        xs[c_off * 725 + l] = (c < CC) ? x[((size_t)b * LL + l) * CC + c] : 0.f;
    }
    __syncthreads();

    for (int r = 0; r < 16; ++r) {
        int c = c0 + r;
        if (c >= CC) break;                       // uniform across block
        size_t nrow = (size_t)(b * CC + c) * LP;
        const float* xr = &xs[r * 725];
        for (int l = tid; l < LP; l += 256) {     // coalesced f16 writes along l
            float tr = 0.f, se = 0.f;
            if (l < LL) {
                float s = 0.f;
#pragma unroll
                for (int w = -12; w <= 12; ++w) {
                    int j = l + w;
                    j = j < 0 ? 0 : (j > LL - 1 ? LL - 1 : j);
                    s += xr[j];
                }
                tr = s * (1.0f / 25.0f);
                se = xr[l] - tr;
            }
            seas[nrow + l]  = (_Float16)se;
            trend[nrow + l] = (_Float16)tr;
        }
    }
}

// ---- GEMM1: r = sigmoid(X[N,736] * W1h^T + b1) -> r256[:,128:] ------------
// 32-row blocks: each wave holds 2 accumulators, B fragment reused 2x.

__global__ __launch_bounds__(256) void gemm1_sigmoid_kernel(
    const _Float16* __restrict__ A,    // [NN, LP]
    const _Float16* __restrict__ W1h,  // [HID_, LP]
    const float* __restrict__ b1,      // [HID_]
    _Float16* __restrict__ r256) {     // [NN, 256]
    __shared__ alignas(16) _Float16 As[32 * 744];
    const int tid = threadIdx.x, lane = tid & 31, wave = tid >> 5;
    const int rowBase = blockIdx.x * 32;

    for (int idx = tid; idx < 32 * 92; idx += 256) {   // 16B chunks
        int r = idx / 92, kc = (idx - r * 92) * 8;
        copy16_g2l(&As[r * 744 + kc], &A[(size_t)(rowBase + r) * LP + kc]);
    }
    staging_fence();

    const int h0 = wave * 16;                     // 8 waves cover HID=128
    float bias = b1[h0 + (lane & 15)];
    float8 c0, c1;
#pragma unroll
    for (int i = 0; i < 8; ++i) { c0[i] = bias; c1[i] = bias; }

    for (int kt = 0; kt < LP / 32; ++kt) {        // 23 iters, 2 WMMAs each
        half16 b  = load_b_frag_T(W1h, LP, h0, kt * 32, lane);
        half16 a0 = load_a_frag(As, 744, 0,  kt * 32, lane);
        half16 a1 = load_a_frag(As, 744, 16, kt * 32, lane);
        c0 = __builtin_amdgcn_wmma_f32_16x16x32_f16(false, a0, false, b,
                                                    (short)0, c0, false, false);
        c1 = __builtin_amdgcn_wmma_f32_16x16x32_f16(false, a1, false, b,
                                                    (short)0, c1, false, false);
    }
#pragma unroll
    for (int rr = 0; rr < 8; ++rr) {
        int m = rr + ((lane >> 4) << 3);
        size_t col = 128 + h0 + (lane & 15);
        r256[(size_t)(rowBase + m) * 256 + col]      = (_Float16)fast_sigmoid(c0[rr]);
        r256[(size_t)(rowBase + 16 + m) * 256 + col] = (_Float16)fast_sigmoid(c1[rr]);
    }
}

// ---------- fused attention: r256[:, :128] = softmax(r mem^T) mem ----------

__global__ __launch_bounds__(256) void attn_kernel(
    _Float16* __restrict__ r256,        // [NN,256] reads cols 128.., writes 0..127
    const _Float16* __restrict__ memh) {// [MM, HID_]
    __shared__ alignas(16) _Float16 rA[16 * 136];
    __shared__ float    logits[16 * 520];
    __shared__ _Float16 att[16 * 520];
    const int tid = threadIdx.x, lane = tid & 31, wave = tid >> 5;
    const int rowBase = blockIdx.x * 16;

    // stage r tile (16 rows x 16 chunks of 16B)
    for (int idx = tid; idx < 16 * 16; idx += 256) {
        int r = idx >> 4, kc = (idx & 15) * 8;
        copy16_g2l(&rA[r * 136 + kc],
                   &r256[(size_t)(rowBase + r) * 256 + 128 + kc]);
    }
    staging_fence();

    // logits[16,512] = r * mem^T   (B(k,f) = mem[f,k] -> contiguous loads)
#pragma unroll
    for (int i = 0; i < 4; ++i) {
        int f0 = (wave * 4 + i) * 16;
        float8 c;
#pragma unroll
        for (int rr = 0; rr < 8; ++rr) c[rr] = 0.f;
#pragma unroll
        for (int kt = 0; kt < 4; ++kt) {
            half16 a = load_a_frag(rA, 136, 0, kt * 32, lane);
            half16 b = load_b_frag_T(memh, HID_, f0, kt * 32, lane);
            c = __builtin_amdgcn_wmma_f32_16x16x32_f16(false, a, false, b,
                                                       (short)0, c, false, false);
        }
#pragma unroll
        for (int rr = 0; rr < 8; ++rr) {
            int m = rr + ((lane >> 4) << 3);
            logits[m * 520 + f0 + (lane & 15)] = c[rr];
        }
    }
    __syncthreads();

    // softmax over F=512 per row; wave handles 2 rows, 16 elems per lane
#pragma unroll
    for (int rsub = 0; rsub < 2; ++rsub) {
        int row = wave * 2 + rsub;
        float vals[16];
        float mx = -3.0e38f;
#pragma unroll
        for (int j = 0; j < 16; ++j) {
            vals[j] = logits[row * 520 + lane + 32 * j];
            mx = fmaxf(mx, vals[j]);
        }
        for (int msk = 16; msk; msk >>= 1) mx = fmaxf(mx, __shfl_xor(mx, msk, 32));
        float s = 0.f;
#pragma unroll
        for (int j = 0; j < 16; ++j) { vals[j] = __expf(vals[j] - mx); s += vals[j]; }
        for (int msk = 16; msk; msk >>= 1) s += __shfl_xor(s, msk, 32);
        float inv = __builtin_amdgcn_rcpf(s);
#pragma unroll
        for (int j = 0; j < 16; ++j)
            att[row * 520 + lane + 32 * j] = (_Float16)(vals[j] * inv);
    }
    __syncthreads();

    // o[16,128] = att[16,512] * mem[512,128]; 8 waves cover 128 cols
    {
        int n0 = wave * 16;
        float8 c;
#pragma unroll
        for (int rr = 0; rr < 8; ++rr) c[rr] = 0.f;
        for (int kt = 0; kt < MM / 32; ++kt) {   // 16 WMMAs
            half16 a = load_a_frag(att, 520, 0, kt * 32, lane);
            half16 b = load_b_frag_N(memh, HID_, kt * 32, n0, lane);
            c = __builtin_amdgcn_wmma_f32_16x16x32_f16(false, a, false, b,
                                                       (short)0, c, false, false);
        }
#pragma unroll
        for (int rr = 0; rr < 8; ++rr) {
            int m = rr + ((lane >> 4) << 3);
            r256[(size_t)(rowBase + m) * 256 + n0 + (lane & 15)] = (_Float16)c[rr];
        }
    }
}

// ---- GEMM2: out_acc[N,336] (+)= r256[N,256] * W2h^T + b2 ------------------
// 32-row blocks, 2 accumulators per wave.

__global__ __launch_bounds__(256) void gemm2_kernel(
    const _Float16* __restrict__ r256,  // [NN,256]
    const _Float16* __restrict__ W2h,   // [PP,256]
    const float* __restrict__ b2,       // [PP]
    float* __restrict__ out_acc,        // [NN, PP]
    int addPrev) {
    __shared__ alignas(16) _Float16 rB[32 * 264];
    const int tid = threadIdx.x, lane = tid & 31, wave = tid >> 5;
    const int rowBase = blockIdx.x * 32;

    for (int idx = tid; idx < 32 * 32; idx += 256) {   // 16B chunks
        int r = idx >> 5, kc = (idx & 31) * 8;
        copy16_g2l(&rB[r * 264 + kc], &r256[(size_t)(rowBase + r) * 256 + kc]);
    }
    staging_fence();

    const int colTile = blockIdx.y * 8 + wave;    // 21 tiles cover P=336
    if (colTile >= PP / 16) return;
    const int p0 = colTile * 16;

    float bias = b2[p0 + (lane & 15)];
    float8 c0, c1;
#pragma unroll
    for (int rr = 0; rr < 8; ++rr) {
        int m = rr + ((lane >> 4) << 3);
        float p0v = 0.f, p1v = 0.f;
        if (addPrev) {
            p0v = out_acc[(size_t)(rowBase + m) * PP + p0 + (lane & 15)];
            p1v = out_acc[(size_t)(rowBase + 16 + m) * PP + p0 + (lane & 15)];
        }
        c0[rr] = bias + p0v;
        c1[rr] = bias + p1v;
    }
    for (int kt = 0; kt < 8; ++kt) {
        half16 b  = load_b_frag_T(W2h, 256, p0, kt * 32, lane);
        half16 a0 = load_a_frag(rB, 264, 0,  kt * 32, lane);
        half16 a1 = load_a_frag(rB, 264, 16, kt * 32, lane);
        c0 = __builtin_amdgcn_wmma_f32_16x16x32_f16(false, a0, false, b,
                                                    (short)0, c0, false, false);
        c1 = __builtin_amdgcn_wmma_f32_16x16x32_f16(false, a1, false, b,
                                                    (short)0, c1, false, false);
    }
#pragma unroll
    for (int rr = 0; rr < 8; ++rr) {
        int m = rr + ((lane >> 4) << 3);
        out_acc[(size_t)(rowBase + m) * PP + p0 + (lane & 15)]      = c0[rr];
        out_acc[(size_t)(rowBase + 16 + m) * PP + p0 + (lane & 15)] = c1[rr];
    }
}

// ---------- final transpose [B,C,P] -> [B,P,C] -----------------------------

__global__ void transpose_out_kernel(const float* __restrict__ acc,
                                     float* __restrict__ out) {
    size_t idx = (size_t)blockIdx.x * blockDim.x + threadIdx.x;
    const size_t total = (size_t)BB * PP * CC;
    if (idx >= total) return;
    int b   = (int)(idx / ((size_t)PP * CC));
    int rem = (int)(idx % ((size_t)PP * CC));
    int p = rem / CC;
    int c = rem - p * CC;
    out[idx] = acc[((size_t)b * CC + c) * PP + p];
}

// ---------------------------- launcher -------------------------------------

extern "C" void kernel_launch(void* const* d_in, const int* in_sizes, int n_in,
                              void* d_out, int out_size, void* d_ws, size_t ws_size,
                              hipStream_t stream) {
    (void)in_sizes; (void)n_in; (void)out_size; (void)ws_size;
    const float* x     = (const float*)d_in[0];
    const float* W_s1  = (const float*)d_in[1];
    const float* b_s1  = (const float*)d_in[2];
    const float* W_s2  = (const float*)d_in[3];
    const float* b_s2  = (const float*)d_in[4];
    const float* W_t1  = (const float*)d_in[5];
    const float* b_t1  = (const float*)d_in[6];
    const float* W_t2  = (const float*)d_in[7];
    const float* b_t2  = (const float*)d_in[8];
    const float* mem_S = (const float*)d_in[9];
    const float* mem_T = (const float*)d_in[10];
    float* out = (float*)d_out;

    // workspace carving (256B aligned)
    size_t off = 0;
    char* base = (char*)d_ws;
    auto carve = [&](size_t bytes) -> void* {
        void* p = base + off;
        off += (bytes + 255) & ~(size_t)255;
        return p;
    };
    _Float16* seas_h  = (_Float16*)carve((size_t)NN * LP * 2);
    _Float16* trend_h = (_Float16*)carve((size_t)NN * LP * 2);
    _Float16* W1s_h   = (_Float16*)carve((size_t)HID_ * LP * 2);
    _Float16* W1t_h   = (_Float16*)carve((size_t)HID_ * LP * 2);
    _Float16* memS_h  = (_Float16*)carve((size_t)MM * HID_ * 2);
    _Float16* memT_h  = (_Float16*)carve((size_t)MM * HID_ * 2);
    _Float16* W2s_h   = (_Float16*)carve((size_t)PP * 256 * 2);
    _Float16* W2t_h   = (_Float16*)carve((size_t)PP * 256 * 2);
    _Float16* r256    = (_Float16*)carve((size_t)NN * 256 * 2);
    float*    out_acc = (float*)   carve((size_t)NN * PP * 4);

    auto cvtGrid = [](size_t n) { return (int)((n + 255) / 256); };

    // weight conversions
    pad_convert_f16_kernel<<<cvtGrid((size_t)HID_ * LP), 256, 0, stream>>>(W_s1, W1s_h, HID_, LL, LP);
    pad_convert_f16_kernel<<<cvtGrid((size_t)HID_ * LP), 256, 0, stream>>>(W_t1, W1t_h, HID_, LL, LP);
    pad_convert_f16_kernel<<<cvtGrid((size_t)MM * HID_), 256, 0, stream>>>(mem_S, memS_h, MM, HID_, HID_);
    pad_convert_f16_kernel<<<cvtGrid((size_t)MM * HID_), 256, 0, stream>>>(mem_T, memT_h, MM, HID_, HID_);
    pad_convert_f16_kernel<<<cvtGrid((size_t)PP * 256), 256, 0, stream>>>(W_s2, W2s_h, PP, 256, 256);
    pad_convert_f16_kernel<<<cvtGrid((size_t)PP * 256), 256, 0, stream>>>(W_t2, W2t_h, PP, 256, 256);

    // decomposition
    decompose_kernel<<<BB * CGRP, 256, 0, stream>>>(x, seas_h, trend_h);

    // seasonal branch
    gemm1_sigmoid_kernel<<<NT32, 256, 0, stream>>>(seas_h, W1s_h, b_s1, r256);
    attn_kernel<<<NT16, 256, 0, stream>>>(r256, memS_h);
    gemm2_kernel<<<dim3(NT32, 3), 256, 0, stream>>>(r256, W2s_h, b_s2, out_acc, 0);

    // trend branch (accumulates into out_acc)
    gemm1_sigmoid_kernel<<<NT32, 256, 0, stream>>>(trend_h, W1t_h, b_t1, r256);
    attn_kernel<<<NT16, 256, 0, stream>>>(r256, memT_h);
    gemm2_kernel<<<dim3(NT32, 3), 256, 0, stream>>>(r256, W2t_h, b_t2, out_acc, 1);

    // final transpose to [B, P, C]
    const size_t total = (size_t)BB * PP * CC;
    transpose_out_kernel<<<cvtGrid(total), 256, 0, stream>>>(out_acc, out);
}
